// NeighborEmbedding_own_24824910971046
// MI455X (gfx1250) — compile-verified, matrix operand order checked by reading, not run on hardware
//
#include <hip/hip_runtime.h>

typedef __attribute__((ext_vector_type(16))) _Float16 v16h;
typedef __attribute__((ext_vector_type(8)))  float    v8f;

#define EPS_BN 1e-5f

// ---------------------------------------------------------------------------
// Generic batched WMMA GEMM with fused epilogue, specialized at compile time
// on TRANSA (A accessed column-major) and SUB (B := B1 - B2).
// Out[b, m, n] = epi( sum_k A(m,k) * B(k,n) )
//   epi: +bias[m]; eval-BN over m; ReLU; + res[b,m,n]
// M % 16 == 0, K % 32 == 0, Np % 64 == 0. Block = 128 threads = 4 waves,
// each wave owns one 16x16 output tile.
//
// Fragment layout (v_wmma_f32_16x16x32_f16, wave32):
//   A: lane(0..15)=row m, halves kh=lane/16 pick K {kh*8..kh*8+7, 16+kh*8..}
//   B: mirrored per column n
//   C/D: vgpr r -> m = r + 8*kh, n = lane&15
// ---------------------------------------------------------------------------
template <int TRANSA, int SUB>
__global__ void __launch_bounds__(128) k_gemm_t(
    const float* __restrict__ A, long long sA, int ldA,
    const float* __restrict__ B1, const float* __restrict__ B2,
    long long sB, int ldB,
    const float* __restrict__ bias,
    const float* __restrict__ bng, const float* __restrict__ bnb,
    const float* __restrict__ bnm, const float* __restrict__ bnv,
    const float* __restrict__ res, long long sR,
    float* __restrict__ Out, long long sO,
    int M, int K, int Np, int doRelu)
{
  const int b    = blockIdx.z;
  const int wave = threadIdx.x >> 5;
  const int lane = threadIdx.x & 31;
  const int n0   = (blockIdx.x * 4 + wave) * 16;
  const int m0   = blockIdx.y * 16;
  if (n0 >= Np) return;                      // wave-uniform
  const int l16 = lane & 15, kh = lane >> 4;

  v8f acc = {0.f, 0.f, 0.f, 0.f, 0.f, 0.f, 0.f, 0.f};

  const float* bp = B1 + b * sB + (n0 + l16) + (size_t)(kh * 8) * ldB;
  const float* cp = SUB ? (B2 + b * sB + (n0 + l16) + (size_t)(kh * 8) * ldB)
                        : nullptr;

  if (TRANSA) {
    // A(m,k) = A[k*ldA + m]; same striding pattern as B.
    const float* ap = A + b * sA + (m0 + l16) + (size_t)(kh * 8) * ldA;
    for (int k0 = 0; k0 < K; k0 += 32) {
      v16h af, bf;
#pragma unroll
      for (int e = 0; e < 8; ++e) {
        af[e]     = (_Float16)ap[(size_t)e * ldA];
        af[e + 8] = (_Float16)ap[(size_t)(16 + e) * ldA];
        float blo = bp[(size_t)e * ldB];
        float bhi = bp[(size_t)(16 + e) * ldB];
        bf[e]     = (_Float16)blo;
        bf[e + 8] = (_Float16)bhi;
      }
      acc = __builtin_amdgcn_wmma_f32_16x16x32_f16(
          false, af, false, bf, (short)0, acc, false, false);
      ap += (size_t)32 * ldA;
      bp += (size_t)32 * ldB;
    }
  } else {
    // A(m,k) = A[m*ldA + k]; two contiguous 8-float runs -> 4x b128 loads.
    const float* ap = A + b * sA + (size_t)(m0 + l16) * ldA + kh * 8;
    for (int k0 = 0; k0 < K; k0 += 32) {
      const float4 a0 = *(const float4*)(ap + 0);
      const float4 a1 = *(const float4*)(ap + 4);
      const float4 a2 = *(const float4*)(ap + 16);
      const float4 a3 = *(const float4*)(ap + 20);
      v16h af, bf;
      af[0]  = (_Float16)a0.x; af[1]  = (_Float16)a0.y;
      af[2]  = (_Float16)a0.z; af[3]  = (_Float16)a0.w;
      af[4]  = (_Float16)a1.x; af[5]  = (_Float16)a1.y;
      af[6]  = (_Float16)a1.z; af[7]  = (_Float16)a1.w;
      af[8]  = (_Float16)a2.x; af[9]  = (_Float16)a2.y;
      af[10] = (_Float16)a2.z; af[11] = (_Float16)a2.w;
      af[12] = (_Float16)a3.x; af[13] = (_Float16)a3.y;
      af[14] = (_Float16)a3.z; af[15] = (_Float16)a3.w;
#pragma unroll
      for (int e = 0; e < 8; ++e) {
        float blo = bp[(size_t)e * ldB];
        float bhi = bp[(size_t)(16 + e) * ldB];
        if (SUB) {
          blo -= cp[(size_t)e * ldB];
          bhi -= cp[(size_t)(16 + e) * ldB];
        }
        bf[e]     = (_Float16)blo;
        bf[e + 8] = (_Float16)bhi;
      }
      acc = __builtin_amdgcn_wmma_f32_16x16x32_f16(
          false, af, false, bf, (short)0, acc, false, false);
      ap += 32;
      bp += (size_t)32 * ldB;
      if (SUB) cp += (size_t)32 * ldB;
    }
  }

  // ---- epilogue: row m = m0 + r + 8*kh, col n = n0 + l16 ----
  const int mb = m0 + 8 * kh;
  float* ob = Out + b * sO + (size_t)mb * Np + (n0 + l16);
  const float* rb = res ? (res + b * sR + (size_t)mb * Np + (n0 + l16)) : nullptr;
#pragma unroll
  for (int r = 0; r < 8; ++r) {
    const int m = mb + r;
    float v = acc[r];
    if (bias) v += bias[m];
    if (bng) {
      const float a = bng[m] * rsqrtf(bnv[m] + EPS_BN);
      v = (v - bnm[m]) * a + bnb[m];
    }
    if (doRelu) v = fmaxf(v, 0.f);
    if (rb) v += rb[(size_t)r * Np];
    ob[(size_t)r * Np] = v;
  }
}

// ---------------------------------------------------------------------------
// SG grouped conv: one block per (batch, center). Builds 2C x 32 matrix
// [f_nb - f_ctr ; f_ctr] in LDS (f16), GEMM with w1 [256 x 2C] via WMMA,
// BN+ReLU, max-pool over the 32 neighbor columns (k<=32, padded w/ center;
// the center is always its own nearest neighbor so padding is max-neutral).
// ---------------------------------------------------------------------------
__global__ void __launch_bounds__(256) k_sg(
    const float* __restrict__ feats, int C, int Nin,
    const int* __restrict__ fid, const int* __restrict__ kid,
    const float* __restrict__ w1,
    const float* __restrict__ bng, const float* __restrict__ bnb,
    const float* __restrict__ bnm, const float* __restrict__ bnv,
    float* __restrict__ out, int s)
{
  const int b = blockIdx.y, sc = blockIdx.x, tid = threadIdx.x;
  __shared__ _Float16 Bm[512][32];   // 2C <= 512
  __shared__ int   nb[32];
  __shared__ float fc[256];          // C <= 256

  const int ci = fid[b * s + sc];
  if (tid < 32) nb[tid] = kid[((long long)b * s + sc) * 32 + tid];
  __syncthreads();

  const float* F = feats + (long long)b * C * Nin;
  for (int c = tid; c < C; c += 256) fc[c] = F[(size_t)c * Nin + ci];
  __syncthreads();

  for (int idx = tid; idx < C * 32; idx += 256) {
    const int c = idx >> 5, j = idx & 31;
    const float fn  = F[(size_t)c * Nin + nb[j]];
    const float fcv = fc[c];
    Bm[c][j]     = (_Float16)(fn - fcv);
    Bm[C + c][j] = (_Float16)fcv;
  }
  __syncthreads();

  const int wave = tid >> 5, lane = tid & 31;
  const int l16 = lane & 15, kh = lane >> 4;
  const int K2 = 2 * C;

  for (int mt = 0; mt < 2; ++mt) {
    const int m0 = (wave + 8 * mt) * 16;        // O = 256 -> 16 M-tiles
    v8f acc0 = {0.f, 0.f, 0.f, 0.f, 0.f, 0.f, 0.f, 0.f};
    v8f acc1 = {0.f, 0.f, 0.f, 0.f, 0.f, 0.f, 0.f, 0.f};
    const float* ap = w1 + (size_t)(m0 + l16) * K2 + kh * 8;
    for (int k0 = 0; k0 < K2; k0 += 32) {
      const float4 a0 = *(const float4*)(ap + 0);
      const float4 a1 = *(const float4*)(ap + 4);
      const float4 a2 = *(const float4*)(ap + 16);
      const float4 a3 = *(const float4*)(ap + 20);
      v16h af, b0, b1;
      af[0]  = (_Float16)a0.x; af[1]  = (_Float16)a0.y;
      af[2]  = (_Float16)a0.z; af[3]  = (_Float16)a0.w;
      af[4]  = (_Float16)a1.x; af[5]  = (_Float16)a1.y;
      af[6]  = (_Float16)a1.z; af[7]  = (_Float16)a1.w;
      af[8]  = (_Float16)a2.x; af[9]  = (_Float16)a2.y;
      af[10] = (_Float16)a2.z; af[11] = (_Float16)a2.w;
      af[12] = (_Float16)a3.x; af[13] = (_Float16)a3.y;
      af[14] = (_Float16)a3.z; af[15] = (_Float16)a3.w;
#pragma unroll
      for (int e = 0; e < 8; ++e) {
        const int klo = k0 + kh * 8 + e;
        const int khi = k0 + 16 + kh * 8 + e;
        b0[e]     = Bm[klo][l16];
        b0[e + 8] = Bm[khi][l16];
        b1[e]     = Bm[klo][16 + l16];
        b1[e + 8] = Bm[khi][16 + l16];
      }
      acc0 = __builtin_amdgcn_wmma_f32_16x16x32_f16(
          false, af, false, b0, (short)0, acc0, false, false);
      acc1 = __builtin_amdgcn_wmma_f32_16x16x32_f16(
          false, af, false, b1, (short)0, acc1, false, false);
      ap += 32;
    }
#pragma unroll
    for (int r = 0; r < 8; ++r) {
      const int m = m0 + r + 8 * kh;
      const float a = bng[m] * rsqrtf(bnv[m] + EPS_BN);
      float v0 = fmaxf((acc0[r] - bnm[m]) * a + bnb[m], 0.f);
      float v1 = fmaxf((acc1[r] - bnm[m]) * a + bnb[m], 0.f);
      float v = fmaxf(v0, v1);
#pragma unroll
      for (int d = 8; d >= 1; d >>= 1) v = fmaxf(v, __shfl_xor(v, d, 32));
      if (l16 == 0) out[(long long)b * 256 * s + (size_t)m * s + sc] = v;
    }
  }
}

// --------------------------- FPS: one block per batch -----------------------
__global__ void __launch_bounds__(512) k_fps(
    const float* __restrict__ xyz, int Nin, int s, int* __restrict__ fid)
{
  const int b = blockIdx.x, tid = threadIdx.x;
  __shared__ float rv[512];
  __shared__ int   ri[512];
  __shared__ float cpt[3];
  const float* P = xyz + (long long)b * Nin * 3;
  const bool act = tid < Nin;
  float px = 0.f, py = 0.f, pz = 0.f, dist = 1e10f;
  if (act) { px = P[tid * 3]; py = P[tid * 3 + 1]; pz = P[tid * 3 + 2]; }
  int far = 0;
  for (int i = 0; i < s; ++i) {
    if (tid == 0) {
      fid[b * s + i] = far;
      cpt[0] = P[far * 3]; cpt[1] = P[far * 3 + 1]; cpt[2] = P[far * 3 + 2];
    }
    __syncthreads();
    if (act) {
      const float dx = px - cpt[0], dy = py - cpt[1], dz = pz - cpt[2];
      dist = fminf(dist, dx * dx + dy * dy + dz * dz);
      rv[tid] = dist; ri[tid] = tid;
    } else { rv[tid] = -1.f; ri[tid] = 0x7fffffff; }
    __syncthreads();
    for (int st = 256; st >= 1; st >>= 1) {
      if (tid < st) {
        const float o = rv[tid + st]; const int oi = ri[tid + st];
        if (o > rv[tid] || (o == rv[tid] && oi < ri[tid])) { rv[tid] = o; ri[tid] = oi; }
      }
      __syncthreads();
    }
    far = ri[0];
    __syncthreads();
  }
}

// --------------------------- kNN: block per (b, center) ---------------------
__global__ void __launch_bounds__(128) k_knn(
    const float* __restrict__ xyz, int Nin, const int* __restrict__ fid,
    int s, int k, int* __restrict__ kid)
{
  const int b = blockIdx.y, sc = blockIdx.x, tid = threadIdx.x;
  __shared__ float d2[512];
  __shared__ float rv[128];
  __shared__ int   ri[128];
  const float* P = xyz + (long long)b * Nin * 3;
  const int ci = fid[b * s + sc];
  const float cx = P[ci * 3], cy = P[ci * 3 + 1], cz = P[ci * 3 + 2];
  for (int n = tid; n < Nin; n += 128) {
    const float dx = P[n * 3] - cx, dy = P[n * 3 + 1] - cy, dz = P[n * 3 + 2] - cz;
    d2[n] = dx * dx + dy * dy + dz * dz;
  }
  __syncthreads();
  int* row = kid + ((long long)b * s + sc) * 32;
  for (int it = 0; it < 32; ++it) {
    if (it < k) {
      float best = 1e30f; int bi = 0x7fffffff;
      for (int n = tid; n < Nin; n += 128) {
        const float v = d2[n];
        if (v < best || (v == best && n < bi)) { best = v; bi = n; }
      }
      rv[tid] = best; ri[tid] = bi;
      __syncthreads();
      for (int st = 64; st >= 1; st >>= 1) {
        if (tid < st) {
          if (rv[tid + st] < rv[tid] ||
              (rv[tid + st] == rv[tid] && ri[tid + st] < ri[tid])) {
            rv[tid] = rv[tid + st]; ri[tid] = ri[tid + st];
          }
        }
        __syncthreads();
      }
      if (tid == 0) { row[it] = ri[0]; d2[ri[0]] = 1e30f; }
      __syncthreads();
    } else if (tid == 0) {
      row[it] = ci;
    }
  }
}

// --------------------------- softmax over rows (in place) -------------------
__global__ void __launch_bounds__(128) k_softmax_row(float* __restrict__ E, int Nk)
{
  const int tid = threadIdx.x;
  __shared__ float red[128];
  float* row = E + (long long)(blockIdx.y * gridDim.x + blockIdx.x) * Nk;
  float mx = -1e30f;
  for (int m = tid; m < Nk; m += 128) mx = fmaxf(mx, row[m]);
  red[tid] = mx; __syncthreads();
  for (int st = 64; st >= 1; st >>= 1) {
    if (tid < st) red[tid] = fmaxf(red[tid], red[tid + st]);
    __syncthreads();
  }
  mx = red[0]; __syncthreads();
  float sm = 0.f;
  for (int m = tid; m < Nk; m += 128) sm += __expf(row[m] - mx);
  red[tid] = sm; __syncthreads();
  for (int st = 64; st >= 1; st >>= 1) {
    if (tid < st) red[tid] += red[tid + st];
    __syncthreads();
  }
  const float inv = 1.f / red[0];
  for (int m = tid; m < Nk; m += 128) row[m] = __expf(row[m] - mx) * inv;
}

// ------------------- column normalization: att /= (1e-9 + colsum) -----------
__global__ void __launch_bounds__(128) k_colnorm(float* __restrict__ E, int Nq, int Nk)
{
  const int m = blockIdx.x * 128 + threadIdx.x;
  if (m >= Nk) return;
  float* base = E + (long long)blockIdx.y * Nq * Nk;
  float s = 0.f;
  for (int n = 0; n < Nq; ++n) s += base[(size_t)n * Nk + m];
  const float inv = 1.f / (1e-9f + s);
  for (int n = 0; n < Nq; ++n) base[(size_t)n * Nk + m] *= inv;
}

// --------------------------- small utility kernels --------------------------
__global__ void k_x2xyz(const float* __restrict__ x, float* __restrict__ xyz,
                        int N, int total)  // x [B,3,N] -> xyz [B,N,3]
{
  const int i = blockIdx.x * 256 + threadIdx.x;
  if (i >= total) return;
  const int b = i / N, n = i % N;
  const float* xb = x + (long long)b * 3 * N;
  xyz[(long long)i * 3 + 0] = xb[n];
  xyz[(long long)i * 3 + 1] = xb[N + n];
  xyz[(long long)i * 3 + 2] = xb[2 * N + n];
}

__global__ void k_gatherxyz(const float* __restrict__ xyz, const int* __restrict__ fid,
                            float* __restrict__ out, int Nin, int s, int total)
{
  const int i = blockIdx.x * 256 + threadIdx.x;
  if (i >= total) return;
  const int b = i / s;
  const float* p = xyz + ((long long)b * Nin + fid[i]) * 3;
  out[(long long)i * 3 + 0] = p[0];
  out[(long long)i * 3 + 1] = p[1];
  out[(long long)i * 3 + 2] = p[2];
}

__global__ void k_tile2(const float* __restrict__ in, float* __restrict__ out,
                        int Nh, int total)  // out[.., 2Nh] = [in, in]
{
  const int i = blockIdx.x * 256 + threadIdx.x;
  if (i >= total) return;
  const int n = i % (2 * Nh), bc = i / (2 * Nh);
  out[i] = in[(long long)bc * Nh + (n % Nh)];
}

__global__ void k_conv1(const float* __restrict__ x, const float* __restrict__ w,
                        const float* g, const float* bt, const float* mu, const float* vr,
                        float* __restrict__ out, int N, int total)
{
  const int i = blockIdx.x * 256 + threadIdx.x;
  if (i >= total) return;
  const int n = i % N, t = i / N, o = t % 128, b = t / 128;
  const float* xb = x + (long long)b * 3 * N;
  float sv = w[o * 3 + 0] * xb[n] + w[o * 3 + 1] * xb[N + n] + w[o * 3 + 2] * xb[2 * N + n];
  const float a = g[o] * rsqrtf(vr[o] + EPS_BN);
  sv = (sv - mu[o]) * a + bt[o];
  out[i] = fmaxf(sv, 0.f);
}

// ===========================================================================
extern "C" void kernel_launch(void* const* d_in, const int* in_sizes, int n_in,
                              void* d_out, int out_size, void* d_ws, size_t ws_size,
                              hipStream_t stream) {
  (void)in_sizes; (void)n_in; (void)out_size; (void)ws_size;
  const int NB = 32, NPTS = 512, C = 256;

  // -------- unpack inputs (setup_inputs dict order, recursive) --------
  int ip = 0;
  auto nf = [&]() { return (const float*)d_in[ip++]; };
  struct BNp { const float *g, *b, *m, *v; };
  auto rbn = [&]() { BNp p; p.g = nf(); p.b = nf(); p.m = nf(); p.v = nf(); return p; };
  struct SGp { const float* w1; BNp bn; };
  auto rsg = [&]() { SGp s; s.w1 = nf(); s.bn = rbn(); return s; };
  struct OAp { const float *qk, *vw, *vb, *tw, *tb; BNp bn; };
  auto roa = [&]() { OAp o; o.qk = nf(); o.vw = nf(); o.vb = nf();
                     o.tw = nf(); o.tb = nf(); o.bn = rbn(); return o; };

  const float* x = nf();
  const float* c1_w = nf(); BNp bn1 = rbn();
  const float* c2_w = nf(); BNp bn2 = rbn();
  SGp sg0 = rsg(); OAp oa01 = roa(), oa02 = roa();
  SGp sg1 = rsg(); OAp oa11 = roa(), oa12 = roa();
  SGp sg2 = rsg(); OAp oa21 = roa(), oa22 = roa();
  SGp sg3 = rsg();
  OAp oa4 = roa();
  OAp oao1 = roa(), oao2 = roa(), oao3 = roa();

  // -------- workspace carving --------
  size_t off = 0;
  auto alloc = [&](size_t nbytes) -> void* {
    void* p = (char*)d_ws + off;
    off += (nbytes + 255) & ~(size_t)255;
    return p;
  };
  const size_t BIG = (size_t)NB * C * NPTS * sizeof(float);   // 16 MB
  float* xyz0 = (float*)alloc((size_t)NB * 512 * 3 * 4);
  float* xyz1 = (float*)alloc((size_t)NB * 256 * 3 * 4);
  float* xyz2 = (float*)alloc((size_t)NB * 128 * 3 * 4);
  int* fid0 = (int*)alloc((size_t)NB * 512 * 4);
  int* kid0 = (int*)alloc((size_t)NB * 512 * 32 * 4);
  int* fid1 = (int*)alloc((size_t)NB * 256 * 4);
  int* kid1 = (int*)alloc((size_t)NB * 256 * 32 * 4);
  int* fid2 = (int*)alloc((size_t)NB * 128 * 4);
  int* kid2 = (int*)alloc((size_t)NB * 128 * 32 * 4);
  int* fid3 = (int*)alloc((size_t)NB * 64 * 4);
  int* kid3 = (int*)alloc((size_t)NB * 64 * 32 * 4);
  float* T1 = (float*)alloc(BIG);
  float* T2 = (float*)alloc(BIG);
  float* T3 = (float*)alloc(BIG);
  float* PA = (float*)alloc(BIG);
  float* PB = (float*)alloc(BIG);
  float* Pf02 = (float*)alloc(BIG);
  float* Pf12 = (float*)alloc(BIG / 2);
  float* Pf22 = (float*)alloc(BIG / 4);
  float* Ebuf = (float*)alloc((size_t)NB * NPTS * NPTS * sizeof(float)); // 32 MB

  // -------- generic GEMM launcher (picks template specialization) --------
  auto gemm = [&](const float* A, long long sA, int ldA, int transA,
                  const float* B1, const float* B2, long long sB, int ldB,
                  const float* bias, const BNp* bn,
                  const float* res, long long sR,
                  float* Out, long long sO,
                  int M, int K, int Np, int relu) {
    dim3 grid(Np / 64, M / 16, NB);
    const float* g  = bn ? bn->g : nullptr;
    const float* bb = bn ? bn->b : nullptr;
    const float* mm = bn ? bn->m : nullptr;
    const float* vv = bn ? bn->v : nullptr;
    if (transA) {
      k_gemm_t<1, 0><<<grid, 128, 0, stream>>>(A, sA, ldA, B1, nullptr, sB, ldB,
          bias, g, bb, mm, vv, res, sR, Out, sO, M, K, Np, relu);
    } else if (B2) {
      k_gemm_t<0, 1><<<grid, 128, 0, stream>>>(A, sA, ldA, B1, B2, sB, ldB,
          bias, g, bb, mm, vv, res, sR, Out, sO, M, K, Np, relu);
    } else {
      k_gemm_t<0, 0><<<grid, 128, 0, stream>>>(A, sA, ldA, B1, nullptr, sB, ldB,
          bias, g, bb, mm, vv, res, sR, Out, sO, M, K, Np, relu);
    }
  };

  // -------- offset attention (self) --------
  auto runOA = [&](const float* Xin, float* Xout, const OAp& p, int qk, int Np) {
    const long long sX = (long long)C * Np;
    // xq = qk_w @ x -> T1 [B,qk,Np]
    gemm(p.qk, 0, C, 0, Xin, nullptr, sX, Np, nullptr, nullptr, nullptr, 0,
         T1, (long long)qk * Np, qk, C, Np, 0);
    // xv = v_w @ x + v_b -> T2
    gemm(p.vw, 0, C, 0, Xin, nullptr, sX, Np, p.vb, nullptr, nullptr, 0,
         T2, sX, C, C, Np, 0);
    // energy = xq^T xq -> E [B,Np,Np]
    gemm(T1, (long long)qk * Np, Np, 1, T1, nullptr, (long long)qk * Np, Np,
         nullptr, nullptr, nullptr, 0, Ebuf, (long long)Np * Np, Np, qk, Np, 0);
    k_softmax_row<<<dim3(Np, NB), 128, 0, stream>>>(Ebuf, Np);
    k_colnorm<<<dim3((Np + 127) / 128, NB), 128, 0, stream>>>(Ebuf, Np, Np);
    // xr = xv @ att -> T3
    gemm(T2, sX, Np, 0, Ebuf, nullptr, (long long)Np * Np, Np,
         nullptr, nullptr, nullptr, 0, T3, sX, C, Np, Np, 0);
    // out = x + relu(bn(t_w @ (x - xr) + t_b))
    gemm(p.tw, 0, C, 0, Xin, T3, sX, Np, p.tb, &p.bn, Xin, sX,
         Xout, sX, C, C, Np, 1);
  };

  // -------- cross offset attention --------
  auto runOAS = [&](const float* Q, const float* Xin, float* Xout,
                    const OAp& p, int qk, int Np) {
    const long long sX = (long long)C * Np;
    gemm(p.qk, 0, C, 0, Q,   nullptr, sX, Np, nullptr, nullptr, nullptr, 0,
         T1, (long long)qk * Np, qk, C, Np, 0);               // xq
    gemm(p.qk, 0, C, 0, Xin, nullptr, sX, Np, nullptr, nullptr, nullptr, 0,
         T2, (long long)qk * Np, qk, C, Np, 0);               // xk
    gemm(p.vw, 0, C, 0, Xin, nullptr, sX, Np, p.vb, nullptr, nullptr, 0,
         T3, sX, C, C, Np, 0);                                 // xv
    gemm(T1, (long long)qk * Np, Np, 1, T2, nullptr, (long long)qk * Np, Np,
         nullptr, nullptr, nullptr, 0, Ebuf, (long long)Np * Np, Np, qk, Np, 0);
    k_softmax_row<<<dim3(Np, NB), 128, 0, stream>>>(Ebuf, Np);
    k_colnorm<<<dim3((Np + 127) / 128, NB), 128, 0, stream>>>(Ebuf, Np, Np);
    gemm(T3, sX, Np, 0, Ebuf, nullptr, (long long)Np * Np, Np,
         nullptr, nullptr, nullptr, 0, T1, sX, C, Np, Np, 0);  // xr -> T1
    gemm(p.tw, 0, C, 0, Xin, T1, sX, Np, p.tb, &p.bn, Xin, sX,
         Xout, sX, C, C, Np, 1);
  };

  // -------- sample + group + conv + maxpool --------
  auto runSG = [&](const float* feats, int Cf, int Nin, const float* coordsIn,
                   float* coordsOut, int s, int k, const SGp& p,
                   int* fid, int* kid, float* out) {
    k_fps<<<NB, 512, 0, stream>>>(coordsIn, Nin, s, fid);
    k_knn<<<dim3(s, NB), 128, 0, stream>>>(coordsIn, Nin, fid, s, k, kid);
    if (coordsOut) {
      const int tot = NB * s;
      k_gatherxyz<<<(tot + 255) / 256, 256, 0, stream>>>(coordsIn, fid, coordsOut,
                                                         Nin, s, tot);
    }
    k_sg<<<dim3(s, NB), 256, 0, stream>>>(feats, Cf, Nin, fid, kid, p.w1,
                                          p.bn.g, p.bn.b, p.bn.m, p.bn.v, out, s);
  };

  // ============================ forward pass ============================
  {
    const int tot = NB * NPTS;
    k_x2xyz<<<(tot + 255) / 256, 256, 0, stream>>>(x, xyz0, NPTS, tot);
  }
  {
    const int tot = NB * 128 * NPTS;
    k_conv1<<<(tot + 255) / 256, 256, 0, stream>>>(x, c1_w, bn1.g, bn1.b, bn1.m, bn1.v,
                                                   T1, NPTS, tot);  // f [B,128,512]
  }
  // f = relu(bn2(c2_w @ f)) -> T2
  gemm(c2_w, 0, 128, 0, T1, nullptr, (long long)128 * NPTS, NPTS,
       nullptr, &bn2, nullptr, 0, T2, (long long)128 * NPTS, 128, 128, NPTS, 1);

  // Stage 0: s=512, k=24, feats 128ch  -> f0 in PA [B,256,512]
  runSG(T2, 128, 512, xyz0, nullptr, 512, 24, sg0, fid0, kid0, PA);
  runOA(PA, PB, oa01, 64, 512);
  runOA(PB, Pf02, oa02, 64, 512);

  // Stage 1: s=256, k=32 (coords: original xyz) -> PA [B,256,256]
  runSG(Pf02, 256, 512, xyz0, xyz1, 256, 32, sg1, fid1, kid1, PA);
  runOA(PA, PB, oa11, 64, 256);
  runOA(PB, Pf12, oa12, 64, 256);

  // Stage 2: s=128, k=32 (coords: xyz1) -> PA [B,256,128]
  runSG(Pf12, 256, 256, xyz1, xyz2, 128, 32, sg2, fid2, kid2, PA);
  runOA(PA, PB, oa21, 64, 128);
  runOA(PB, Pf22, oa22, 64, 128);

  // Stage 3: s=64, k=32 (coords: xyz2) -> PA [B,256,64]
  runSG(Pf22, 256, 128, xyz2, nullptr, 64, 32, sg3, fid3, kid3, PA);
  runOA(PA, PB, oa21, 64, 64);   // source reuses oa21/oa22 weights
  runOA(PB, PA, oa22, 64, 64);
  runOA(PA, PB, oa4, 64, 64);    // y [B,256,64] in PB

  // decoder: concat + cross attention
  {
    const int tot = NB * C * 128;
    k_tile2<<<(tot + 255) / 256, 256, 0, stream>>>(PB, PA, 64, tot);
  }
  runOAS(PA, Pf22, PB, oao1, 256, 128);
  {
    const int tot = NB * C * 256;
    k_tile2<<<(tot + 255) / 256, 256, 0, stream>>>(PB, PA, 128, tot);
  }
  runOAS(PA, Pf12, PB, oao2, 256, 256);
  {
    const int tot = NB * C * 512;
    k_tile2<<<(tot + 255) / 256, 256, 0, stream>>>(PB, PA, 256, tot);
  }
  runOAS(PA, Pf02, (float*)d_out, oao3, 256, 512);   // [B,256,512]
}
